// ConditionalBlock_15882789060921
// MI455X (gfx1250) — matrix operand order repaired and checked by reading, not compiled, not run on hardware
//
#include <hip/hip_runtime.h>
#include <math.h>
#include <stdint.h>

typedef __bf16 bf16;
typedef __attribute__((ext_vector_type(8)))  __bf16 v8bf;
typedef __attribute__((ext_vector_type(16))) __bf16 v16bf;
typedef __attribute__((ext_vector_type(8)))  float  v8f;

#define ZV8 {0.f,0.f,0.f,0.f,0.f,0.f,0.f,0.f}

// ---------------------------------------------------------------- WMMA helper
__device__ __forceinline__ v8f wmma_bf16(v16bf a, v16bf b, v8f c) {
#if defined(__HIP_DEVICE_COMPILE__)
  return __builtin_amdgcn_wmma_f32_16x16x32_bf16(false, a, false, b, (short)0, c, false, false);
#else
  return c;
#endif
}

// A-operand fragment (16-bit 16x32): caller passes p = base + (lane&15)*stride,
// b0 = (lane>>4)*8.  Lane holds K chunks [b0..b0+7] and [16+b0..16+b0+7].
__device__ __forceinline__ v16bf frag_a(const bf16* p, int b0) {
  union { v16bf v; v8bf h[2]; } u;
  u.h[0] = *(const v8bf*)(p + b0);
  u.h[1] = *(const v8bf*)(p + 16 + b0);
  return u.v;
}
// B-operand fragment (16-bit 32x16, column-per-lane): caller passes p = colbase,
// k0 = (lane>>4)*16.  Lane holds K chunks [k0/2..k0/2+7] and [16+k0/2..16+k0/2+7].
__device__ __forceinline__ v16bf frag_b(const bf16* p, int k0) {
  union { v16bf v; v8bf h[2]; } u;
  u.h[0] = *(const v8bf*)(p + (k0 >> 1));
  u.h[1] = *(const v8bf*)(p + 16 + (k0 >> 1));
  return u.v;
}

// ---------------------------------------------------------------- async-to-LDS
// Low 32 bits of a generic pointer to LDS = group-segment byte offset.
__device__ __forceinline__ unsigned lds_lo32(const void* p) {
  return (unsigned)(unsigned long long)(uintptr_t)p;
}
// 16B global -> LDS async copy (ASYNCcnt-tracked, no VGPR staging).
__device__ __forceinline__ void async_copy_b128(unsigned lds_off, const bf16* g) {
  asm volatile("global_load_async_to_lds_b128 %0, %1, off"
               :: "v"(lds_off), "v"((unsigned long long)(uintptr_t)g)
               : "memory");
}
template <int N>
__device__ __forceinline__ void wait_asynccnt() {
  asm volatile("s_wait_asynccnt %0" :: "i"(N) : "memory");
}

// ---------------------------------------------------------------- small kernels
// mod = silu(c) @ W_mod + b_mod   -> [4, 6144] f32
__global__ __launch_bounds__(256)
void mod_kernel(const float* __restrict__ c, const float* __restrict__ Wm,
                const float* __restrict__ bm, float* __restrict__ modv) {
  const int idx = blockIdx.x * 256 + threadIdx.x;      // 0..24575
  const int b = idx / 6144, j = idx % 6144;
  const float* cb = c + b * 1024;
  float acc = bm[j];
  for (int k = 0; k < 1024; ++k) {
    const float cv = cb[k];
    const float s = cv / (1.f + __expf(-cv));          // silu
    acc = fmaf(s, Wm[(size_t)k * 6144 + j], acc);
  }
  modv[idx] = acc;
}

// Wt[n*K + k] = (bf16) W[k*N + n]
__global__ __launch_bounds__(256)
void cvt_t_kernel(const float* __restrict__ W, bf16* __restrict__ Wt, int K, int N) {
  const size_t idx = (size_t)blockIdx.x * 256 + threadIdx.x;
  if (idx >= (size_t)K * N) return;
  const int k = (int)(idx / N), n = (int)(idx % N);
  Wt[(size_t)n * K + k] = (bf16)W[idx];
}

__device__ __forceinline__ float block_sum(float v, float* red) {
  #pragma unroll
  for (int m = 16; m >= 1; m >>= 1) v += __shfl_xor(v, m, 32);
  if ((threadIdx.x & 31) == 0) red[threadIdx.x >> 5] = v;
  __syncthreads();
  float t = 0.f;
  #pragma unroll
  for (int i = 0; i < 8; ++i) t += red[i];
  __syncthreads();
  return t;
}

// per-row affine-free LayerNorm + AdaLN modulate, f32 -> bf16
__global__ __launch_bounds__(256)
void lnmod_kernel(const float* __restrict__ x, const float* __restrict__ modv,
                  int sh_off, int sc_off, bf16* __restrict__ out) {
  __shared__ float red[8];
  const int row = blockIdx.x, b = row >> 10;
  const float* xr = x + (size_t)row * 1024;
  float xv[4];
  #pragma unroll
  for (int i = 0; i < 4; ++i) xv[i] = xr[threadIdx.x + 256 * i];
  const float mu = block_sum(xv[0] + xv[1] + xv[2] + xv[3], red) * (1.f / 1024.f);
  float vs = 0.f;
  #pragma unroll
  for (int i = 0; i < 4; ++i) { const float d = xv[i] - mu; vs = fmaf(d, d, vs); }
  const float var = block_sum(vs, red) * (1.f / 1024.f);
  const float rstd = rsqrtf(var + 1e-5f);
  const float* shv = modv + b * 6144 + sh_off;
  const float* scv = modv + b * 6144 + sc_off;
  #pragma unroll
  for (int i = 0; i < 4; ++i) {
    const int col = threadIdx.x + 256 * i;
    const float y = (xv[i] - mu) * rstd;
    out[(size_t)row * 1024 + col] = (bf16)(y * (1.f + scv[col]) + shv[col]);
  }
}

// ---------------------------------------------------------------- WMMA GEMM
// C[M,N] = A[M,K](bf16) * Bt[N,K](bf16)^T ; async double-buffered LDS staging.
// EPI 0: QKV scatter (q scaled 0.125, v transposed)  EPI 1: resid+gate+bias f32
// EPI 2: bias + exact gelu -> bf16
struct GemmArgs {
  const bf16* A; const bf16* Bt;
  int M, N, K;
  const float* bias; const float* gvec; const float* resid;
  float* outf; bf16* outb;
  bf16* oq; bf16* ok; bf16* ov;
};

template <int EPI>
__global__ __launch_bounds__(256) void gemm_kernel(GemmArgs g) {
  __shared__ __align__(16) bf16 As[2][128][40];
  __shared__ __align__(16) bf16 Bs[2][128][40];
  const int tid = threadIdx.x;
  const int lane = tid & 31, w = tid >> 5;
  const int wm = w >> 2, wn = w & 3;
  const int m16 = lane & 15, halfq = lane >> 4;
  const int b0 = halfq * 8, k0 = halfq * 16;
  const int bn = blockIdx.x, bm = blockIdx.y;

  const v8f zero = ZV8;
  v8f acc[4][2];
  #pragma unroll
  for (int i = 0; i < 4; ++i)
    #pragma unroll
    for (int j = 0; j < 2; ++j) acc[i][j] = zero;

  const int ldrow = tid >> 1, ldseg = (tid & 1) * 16;
  const bf16* gA = g.A  + (size_t)(bm * 128 + ldrow) * g.K + ldseg;
  const bf16* gB = g.Bt + (size_t)(bn * 128 + ldrow) * g.K + ldseg;
  const unsigned lA[2] = { lds_lo32(&As[0][ldrow][ldseg]), lds_lo32(&As[1][ldrow][ldseg]) };
  const unsigned lB[2] = { lds_lo32(&Bs[0][ldrow][ldseg]), lds_lo32(&Bs[1][ldrow][ldseg]) };

  // prologue: stage k-slice 0 into buffer 0
  async_copy_b128(lA[0],      gA);
  async_copy_b128(lA[0] + 16, gA + 8);
  async_copy_b128(lB[0],      gB);
  async_copy_b128(lB[0] + 16, gB + 8);
  gA += 32; gB += 32;

  const int ksteps = g.K >> 5;
  for (int kk = 0; kk < ksteps; ++kk) {
    const int cur = kk & 1;
    if (kk + 1 < ksteps) {
      const int nxt = cur ^ 1;
      async_copy_b128(lA[nxt],      gA);
      async_copy_b128(lA[nxt] + 16, gA + 8);
      async_copy_b128(lB[nxt],      gB);
      async_copy_b128(lB[nxt] + 16, gB + 8);
      gA += 32; gB += 32;
      wait_asynccnt<4>();          // in-order: current slice complete
    } else {
      wait_asynccnt<0>();
    }
    __syncthreads();
    v16bf af[4], bfr[2];
    #pragma unroll
    for (int i = 0; i < 4; ++i) af[i] = frag_a(&As[cur][wm * 64 + i * 16 + m16][0], b0);
    #pragma unroll
    for (int j = 0; j < 2; ++j) bfr[j] = frag_b(&Bs[cur][wn * 32 + j * 16 + m16][0], k0);
    #pragma unroll
    for (int i = 0; i < 4; ++i)
      #pragma unroll
      for (int j = 0; j < 2; ++j)
        acc[i][j] = wmma_bf16(af[i], bfr[j], acc[i][j]);
    __syncthreads();               // protect buffer before next prefetch
  }

  #pragma unroll
  for (int i = 0; i < 4; ++i) {
    #pragma unroll
    for (int j = 0; j < 2; ++j) {
      #pragma unroll
      for (int r = 0; r < 8; ++r) {
        const int row = bm * 128 + wm * 64 + i * 16 + r + halfq * 8;
        const int col = bn * 128 + wn * 32 + j * 16 + m16;
        const float v = acc[i][j][r];
        if (EPI == 0) {
          const int b = row >> 10, t = row & 1023, d = col & 63;
          if (col < 1024) {
            g.oq[((size_t)(b * 16 + (col >> 6)) * 1024 + t) * 64 + d] = (bf16)(v * 0.125f);
          } else if (col < 2048) {
            g.ok[((size_t)(b * 16 + ((col - 1024) >> 6)) * 1024 + t) * 64 + d] = (bf16)v;
          } else {
            g.ov[((size_t)(b * 16 + ((col - 2048) >> 6)) * 64 + d) * 1024 + t] = (bf16)v;
          }
        } else if (EPI == 1) {
          const int b = row >> 10;
          const size_t off = (size_t)row * g.N + col;
          g.outf[off] = g.resid[off] + g.gvec[b * 6144 + col] * (v + g.bias[col]);
        } else {
          const float t2 = v + g.bias[col];
          const float ge = 0.5f * t2 * (1.f + erff(t2 * 0.70710678118f));
          g.outb[(size_t)row * g.N + col] = (bf16)ge;
        }
      }
    }
  }
}

// ---------------------------------------------------------------- flash attention
// Block = 8 waves = 8 query tiles of ONE (b,h); K/V tiles staged cooperatively
// in double-buffered LDS via async copies (8x traffic reduction vs per-wave).
// q,k: [B,H,T,64] bf16 (q pre-scaled by 1/8);  vT: [B,H,64,T] bf16;
// ao: [B,T,16*64] bf16.
__global__ __launch_bounds__(256)
void attn_kernel(const bf16* __restrict__ q, const bf16* __restrict__ kmat,
                 const bf16* __restrict__ vT, bf16* __restrict__ ao) {
  __shared__ __align__(16) bf16 Ks[2][32][72];   // [key][dh], padded
  __shared__ __align__(16) bf16 Vs[2][64][40];   // [dim][key], padded
  __shared__ __align__(16) bf16 pbuf[8][512];    // per-wave 16x32 prob tile
  const int tid = threadIdx.x;
  const int w = tid >> 5, lane = tid & 31;
  const int m16 = lane & 15, halfq = lane >> 4;
  const int b0 = halfq * 8, k0 = halfq * 16;
  const int g = blockIdx.x * 8 + w;              // 0..4095; same (b,h) per block
  const int b = g >> 10, h = (g >> 6) & 15, qt = g & 63;
  const size_t headoff = (size_t)(b * 16 + h) * 1024 * 64;

  const bf16* qrow = q + headoff + (size_t)(qt * 16 + m16) * 64;
  const v16bf qf0 = frag_a(qrow, b0);            // dh 0..31
  const v16bf qf1 = frag_a(qrow + 32, b0);       // dh 32..63
  const bf16* kh = kmat + headoff;
  const bf16* vh = vT + headoff;

  // cooperative K/V staging addresses (uniform head across the block)
  const int krow = tid >> 3, kchunk = (tid & 7) * 8;   // 32 x 64
  const int vrow = tid >> 2, vchunk = (tid & 3) * 8;   // 64 x 32
  const bf16* gK0 = kh + (size_t)krow * 64 + kchunk;
  const bf16* gV0 = vh + (size_t)vrow * 1024 + vchunk;
  const unsigned lK[2] = { lds_lo32(&Ks[0][krow][kchunk]), lds_lo32(&Ks[1][krow][kchunk]) };
  const unsigned lV[2] = { lds_lo32(&Vs[0][vrow][vchunk]), lds_lo32(&Vs[1][vrow][vchunk]) };

  const v8f zero = ZV8;
  v8f o[4];
  #pragma unroll
  for (int i = 0; i < 4; ++i) o[i] = zero;
  float mrow[8], lrow[8];
  #pragma unroll
  for (int r = 0; r < 8; ++r) { mrow[r] = -3.0e38f; lrow[r] = 0.f; }

  // prologue: stage key-tile 0
  async_copy_b128(lK[0], gK0);
  async_copy_b128(lV[0], gV0);

  for (int kt = 0; kt < 32; ++kt) {
    const int cur = kt & 1;
    if (kt + 1 < 32) {
      async_copy_b128(lK[cur ^ 1], gK0 + (size_t)(kt + 1) * 2048);
      async_copy_b128(lV[cur ^ 1], gV0 + (size_t)(kt + 1) * 32);
      wait_asynccnt<2>();
    } else {
      wait_asynccnt<0>();
    }
    __syncthreads();

    const bf16* k0p = &Ks[cur][m16][0];
    const bf16* k1p = &Ks[cur][16 + m16][0];
    v8f s0 = zero, s1 = zero;
    s0 = wmma_bf16(qf0, frag_b(k0p, k0), s0);
    s0 = wmma_bf16(qf1, frag_b(k0p + 32, k0), s0);
    s1 = wmma_bf16(qf0, frag_b(k1p, k0), s1);
    s1 = wmma_bf16(qf1, frag_b(k1p + 32, k0), s1);

    float alpha[8];
    #pragma unroll
    for (int r = 0; r < 8; ++r) {
      float mx = fmaxf(s0[r], s1[r]);
      mx = fmaxf(mx, __shfl_xor(mx, 1, 32));
      mx = fmaxf(mx, __shfl_xor(mx, 2, 32));
      mx = fmaxf(mx, __shfl_xor(mx, 4, 32));
      mx = fmaxf(mx, __shfl_xor(mx, 8, 32));
      const float mn = fmaxf(mrow[r], mx);
      alpha[r] = __expf(mrow[r] - mn);
      mrow[r] = mn;
      const float p0 = __expf(s0[r] - mn);
      const float p1 = __expf(s1[r] - mn);
      float rs = p0 + p1;
      rs += __shfl_xor(rs, 1, 32);
      rs += __shfl_xor(rs, 2, 32);
      rs += __shfl_xor(rs, 4, 32);
      rs += __shfl_xor(rs, 8, 32);
      lrow[r] = lrow[r] * alpha[r] + rs;
      pbuf[w][(r + 8 * halfq) * 32 + m16]      = (bf16)p0;
      pbuf[w][(r + 8 * halfq) * 32 + 16 + m16] = (bf16)p1;
    }
    #pragma unroll
    for (int i = 0; i < 4; ++i)
      #pragma unroll
      for (int r = 0; r < 8; ++r) o[i][r] *= alpha[r];

    const v16bf pf = frag_a(&pbuf[w][m16 * 32], b0);          // P: 16x32 A-operand
    #pragma unroll
    for (int nb = 0; nb < 4; ++nb)
      o[nb] = wmma_bf16(pf, frag_b(&Vs[cur][nb * 16 + m16][0], k0), o[nb]);

    __syncthreads();               // protect K/V buffers before next prefetch
  }

  #pragma unroll
  for (int r = 0; r < 8; ++r) {
    const float inv = 1.f / lrow[r];
    const int t = qt * 16 + r + 8 * halfq;
    const size_t rowoff = ((size_t)b * 1024 + t) * 1024 + h * 64 + m16;
    ao[rowoff + 0]  = (bf16)(o[0][r] * inv);
    ao[rowoff + 16] = (bf16)(o[1][r] * inv);
    ao[rowoff + 32] = (bf16)(o[2][r] * inv);
    ao[rowoff + 48] = (bf16)(o[3][r] * inv);
  }
}

// ---------------------------------------------------------------- launch
extern "C" void kernel_launch(void* const* d_in, const int* in_sizes, int n_in,
                              void* d_out, int out_size, void* d_ws, size_t ws_size,
                              hipStream_t stream) {
  (void)in_sizes; (void)n_in; (void)out_size;
  const float* x    = (const float*)d_in[0];
  const float* c    = (const float*)d_in[1];
  const float* Wmod = (const float*)d_in[2];
  const float* bmod = (const float*)d_in[3];
  const float* Wqkv = (const float*)d_in[4];
  const float* Wout = (const float*)d_in[5];
  const float* bout = (const float*)d_in[6];
  const float* W1   = (const float*)d_in[7];
  const float* b1   = (const float*)d_in[8];
  const float* W2   = (const float*)d_in[9];
  const float* b2   = (const float*)d_in[10];
  float* out = (float*)d_out;

  char* base = (char*)d_ws;
  size_t off = 0;
  auto alloc = [&](size_t bytes) {
    void* p = base + off;
    off += (bytes + 255) & ~(size_t)255;
    return p;
  };
  float* modv = (float*)alloc((size_t)4 * 6144 * 4);
  bf16* wqkvt = (bf16*)alloc((size_t)3072 * 1024 * 2);
  bf16* woutt = (bf16*)alloc((size_t)1024 * 1024 * 2);
  bf16* w1t   = (bf16*)alloc((size_t)4096 * 1024 * 2);
  bf16* w2t   = (bf16*)alloc((size_t)1024 * 4096 * 2);
  bf16* xm    = (bf16*)alloc((size_t)4096 * 1024 * 2);   // LN1 out; reused as attn out
  bf16* qb    = (bf16*)alloc((size_t)4096 * 1024 * 2);   // q; reused as LN2 out
  bf16* kb    = (bf16*)alloc((size_t)4096 * 1024 * 2);
  bf16* vtb   = (bf16*)alloc((size_t)4096 * 1024 * 2);
  float* x1   = (float*)alloc((size_t)4096 * 1024 * 4);
  bf16* hbuf  = (bf16*)alloc((size_t)4096 * 4096 * 2);
  if (off > ws_size) return;
  bf16* ao  = xm;
  bf16* xm2 = qb;

  // weights -> bf16, transposed to [N,K]
  cvt_t_kernel<<<(1024 * 3072) / 256, 256, 0, stream>>>(Wqkv, wqkvt, 1024, 3072);
  cvt_t_kernel<<<(1024 * 1024) / 256, 256, 0, stream>>>(Wout, woutt, 1024, 1024);
  cvt_t_kernel<<<(1024 * 4096) / 256, 256, 0, stream>>>(W1, w1t, 1024, 4096);
  cvt_t_kernel<<<(4096 * 1024) / 256, 256, 0, stream>>>(W2, w2t, 4096, 1024);

  // AdaLN modulation vector
  mod_kernel<<<(4 * 6144) / 256, 256, 0, stream>>>(c, Wmod, bmod, modv);

  // LN + modulate (MSA)
  lnmod_kernel<<<4096, 256, 0, stream>>>(x, modv, 0, 1024, xm);

  // QKV projection
  {
    GemmArgs ga{xm, wqkvt, 4096, 3072, 1024,
                nullptr, nullptr, nullptr, nullptr, nullptr, qb, kb, vtb};
    gemm_kernel<0><<<dim3(3072 / 128, 4096 / 128), 256, 0, stream>>>(ga);
  }

  // flash attention
  attn_kernel<<<512, 256, 0, stream>>>(qb, kb, vtb, ao);

  // out projection + gate + residual -> x1
  {
    GemmArgs ga{ao, woutt, 4096, 1024, 1024,
                bout, modv + 2048, x, x1, nullptr, nullptr, nullptr, nullptr};
    gemm_kernel<1><<<dim3(1024 / 128, 4096 / 128), 256, 0, stream>>>(ga);
  }

  // LN + modulate (MLP)
  lnmod_kernel<<<4096, 256, 0, stream>>>(x1, modv, 3072, 4096, xm2);

  // FF1 + gelu
  {
    GemmArgs ga{xm2, w1t, 4096, 4096, 1024,
                b1, nullptr, nullptr, nullptr, hbuf, nullptr, nullptr, nullptr};
    gemm_kernel<2><<<dim3(4096 / 128, 4096 / 128), 256, 0, stream>>>(ga);
  }

  // FF2 + gate + residual -> out
  {
    GemmArgs ga{hbuf, w2t, 4096, 1024, 4096,
                b2, modv + 5120, x1, out, nullptr, nullptr, nullptr, nullptr};
    gemm_kernel<1><<<dim3(1024 / 128, 4096 / 128), 256, 0, stream>>>(ga);
  }
}
